// PCAProjector_37108517437623
// MI455X (gfx1250) — matrix operand order repaired and checked by reading, not compiled, not run on hardware
//
#include <hip/hip_runtime.h>
#include <hip/hip_bf16.h>

// Problem constants (from reference): B=16, S=8192, H=256, nh=4, hs=64
#define B_  16
#define S_  8192
#define H_  256
#define NH_ 4
#define HS_ 64
#define MT_ 128   // rows of x per workgroup

typedef __bf16 bf16_t;
typedef __attribute__((ext_vector_type(16))) __bf16 v16bf;
typedef __attribute__((ext_vector_type(8)))  __bf16 v8bf;
typedef __attribute__((ext_vector_type(4)))  __bf16 v4bf;
typedef __attribute__((ext_vector_type(8)))  float  v8f;
typedef __attribute__((ext_vector_type(4)))  float  v4f;   // native vector for NT loads

union FragU { v16bf v; v8bf h[2]; };

__device__ __forceinline__ v8f wmma_bf16(v16bf a, v16bf b, v8f c) {
  // 8-arg form: (neg_a, A, neg_b, B, c_mod, C, reuse_a, reuse_b)
  return __builtin_amdgcn_wmma_f32_16x16x32_bf16(false, a, false, b,
                                                 (short)0, c, false, false);
}

// ---------------------------------------------------------------------------
// Kernel 0: zero the covariance accumulator (B*NH*HS*HS f32 = 1 MB in d_ws)
// ---------------------------------------------------------------------------
__global__ void k_zero(float* __restrict__ p, int n) {
  int i = blockIdx.x * 256 + threadIdx.x;
  if (i < n) p[i] = 0.0f;
}

// ---------------------------------------------------------------------------
// Kernel 1: fused  h = (x @ W_ff + b_ff)/sqrt(S)   (h never hits HBM)
//           cov[b,head] += h_head^T * h_head   (upper-triangle tiles only,
//                                               f32 atomics into d_ws)
// One WG per (batch b, 128-row slab of S). 256 threads = 8 waves.
// ---------------------------------------------------------------------------
__global__ __launch_bounds__(256) void k_ff_cov(const float* __restrict__ x,
                                                const float* __restrict__ Wff,
                                                const float* __restrict__ bff,
                                                float* __restrict__ cov) {
  __shared__ bf16_t xt[MT_][H_];   // x tile, pre-scaled, bf16      (64 KB)
  __shared__ bf16_t ht[H_][MT_];   // h tile stored TRANSPOSED      (64 KB)

  const float inv_s = 0.011048543456039806f;  // 1/sqrt(8192)
  const int tid  = threadIdx.x;
  const int lane = tid & 31;
  const int wave = tid >> 5;
  const int ln   = lane & 15;   // N index within 16x16 tile
  const int lh   = lane >> 4;   // half-wave selector
  const int b    = blockIdx.y;
  const int s0   = blockIdx.x * MT_;

  // ---- stage x tile (contiguous MT_*H_ floats) -> bf16 LDS, scaled ----
  // x is streamed exactly once -> non-temporal so it doesn't churn L2.
  const float* xbase = x + ((size_t)b * S_ + s0) * H_;
  for (int i = tid; i < MT_ * H_ / 4; i += 256) {
    v4f v = __builtin_nontemporal_load(((const v4f*)xbase) + i);
    int e = i * 4;
    v4bf pk;
    pk[0] = (bf16_t)(v[0] * inv_s);
    pk[1] = (bf16_t)(v[1] * inv_s);
    pk[2] = (bf16_t)(v[2] * inv_s);
    pk[3] = (bf16_t)(v[3] * inv_s);
    *(v4bf*)&xt[e >> 8][e & 255] = pk;
  }
  __syncthreads();

  // ---- phase 1: FF GEMM. Wave w owns output columns [32w, 32w+32). ----
  // B fragments (16-bit B 32x16: lane holds N=ln, 16 consecutive K at lh*16)
  // loaded once from global W (L2-resident) into VGPRs.
  const int nA = wave * 2;
  v16bf bfrag[2][8];
  float bias[2];
  #pragma unroll
  for (int nn = 0; nn < 2; ++nn) {
    const int ncol = (nA + nn) * 16 + ln;
    bias[nn] = bff[ncol] * inv_s;
    #pragma unroll
    for (int ks = 0; ks < 8; ++ks) {
      const int kb = ks * 32 + lh * 16;
      v16bf f;
      #pragma unroll
      for (int j = 0; j < 16; ++j)
        f[j] = (bf16_t)Wff[(size_t)(kb + j) * H_ + ncol];
      bfrag[nn][ks] = f;
    }
  }

  #pragma unroll
  for (int mt = 0; mt < 8; ++mt) {
    v8f acc0 = {}; v8f acc1 = {};
    #pragma unroll
    for (int ks = 0; ks < 8; ++ks) {
      // A fragment (16-bit A 16x32): lane = row M=ln, 16 consecutive K at lh*16
      FragU a;
      const bf16_t* ap = &xt[mt * 16 + ln][ks * 32 + lh * 16];
      a.h[0] = *(const v8bf*)ap;
      a.h[1] = *(const v8bf*)(ap + 8);
      acc0 = wmma_bf16(a.v, bfrag[0][ks], acc0);
      acc1 = wmma_bf16(a.v, bfrag[1][ks], acc1);
    }
    // C layout: lane holds N=ln, M = lh*8 + r (r = vgpr 0..7) -> 8 consecutive
    // s values at fixed feature: store transposed as one 16B chunk.
    const int srow = mt * 16 + lh * 8;
    {
      v8bf hv;
      #pragma unroll
      for (int r = 0; r < 8; ++r) hv[r] = (bf16_t)(acc0[r] + bias[0]);
      *(v8bf*)&ht[nA * 16 + ln][srow] = hv;
    }
    {
      v8bf hv;
      #pragma unroll
      for (int r = 0; r < 8; ++r) hv[r] = (bf16_t)(acc1[r] + bias[1]);
      *(v8bf*)&ht[(nA + 1) * 16 + ln][srow] = hv;
    }
  }
  __syncthreads();

  // ---- phase 2: per-head covariance, UPPER-TRIANGLE tiles only ----
  // cov is symmetric: 10 of 16 (it,kt) tiles per head, 4 heads = 40 jobs,
  // 5 jobs per wave. ht is [feature][s]: all fragment loads contiguous.
  static const signed char TI[10] = {0,0,0,0,1,1,1,2,2,3};
  static const signed char TK[10] = {0,1,2,3,1,2,3,2,3,3};
  #pragma unroll
  for (int j5 = 0; j5 < 5; ++j5) {
    const int job  = wave * 5 + j5;
    const int head = job / 10;
    const int pi   = job - head * 10;
    const int it   = TI[pi];
    const int kt   = TK[pi];
    v8f acc = {};
    #pragma unroll
    for (int ss = 0; ss < 4; ++ss) {
      const int sb = ss * 32;
      FragU a, bb;
      const bf16_t* ap = &ht[head * 64 + it * 16 + ln][sb + lh * 16];
      a.h[0] = *(const v8bf*)ap;
      a.h[1] = *(const v8bf*)(ap + 8);
      const bf16_t* bp = &ht[head * 64 + kt * 16 + ln][sb + lh * 16];
      bb.h[0] = *(const v8bf*)bp;
      bb.h[1] = *(const v8bf*)(bp + 8);
      acc = wmma_bf16(a.v, bb.v, acc);
    }
    float* cb = cov + ((size_t)b * NH_ + head) * HS_ * HS_
                    + (size_t)(it * 16) * HS_ + kt * 16 + ln;
    #pragma unroll
    for (int r = 0; r < 8; ++r)
      atomicAdd(cb + (size_t)(r + lh * 8) * HS_, acc[r]);
  }
}

// ---------------------------------------------------------------------------
// Kernel 2: parallel two-sided Jacobi eigensolver on each 64x64 cov matrix.
// 256 threads/WG: each Jacobi step splits the 32 disjoint rotations into
// 4 groups of 8; thread (r, g) handles group g's column rotations of row r,
// thread (c, g) handles group g's row rotations of column c. All updates
// disjoint. Then ascending rank-sort (eigh convention) and pca @ W_pt into
// d_out. 64 WGs.
// ---------------------------------------------------------------------------
__global__ __launch_bounds__(256) void k_eigh_out(const float* __restrict__ cov,
                                                  const float* __restrict__ Wpt,
                                                  const float* __restrict__ bpt,
                                                  float* __restrict__ out) {
  __shared__ float A[64][65];   // padded: stride 65 kills bank conflicts
  __shared__ float V[64][65];
  __shared__ float rc[32], rs[32];
  __shared__ int   rp[32], rq[32];
  __shared__ float wvals[64];
  __shared__ int   colOf[64];
  __shared__ float wperm[64];

  const int tid = threadIdx.x;
  const int mat = blockIdx.x;           // b*NH + head
  const float* src = cov + (size_t)mat * HS_ * HS_;

  // load + symmetrize (only upper triangle was accumulated)
  for (int idx = tid; idx < 64 * 64; idx += 256) {
    const int i = idx >> 6, j = idx & 63;
    const int lo = i < j ? i : j;
    const int hi = i < j ? j : i;
    A[i][j] = src[lo * 64 + hi];
    V[i][j] = (i == j) ? 1.0f : 0.0f;
  }
  __syncthreads();

  const int rr = tid & 63;    // row (col phase) / column (row phase) owned
  const int gg = tid >> 6;    // pair group 0..3

  for (int sweep = 0; sweep < 12; ++sweep) {
    for (int step = 0; step < 63; ++step) {
      // 32 disjoint pivot pairs (round-robin tournament, element 63 fixed)
      if (tid < 32) {
        int p, q;
        if (tid == 0) { p = 63; q = step; }
        else {
          p = (step + tid) % 63;
          q = (step - tid + 63) % 63;
        }
        if (p > q) { int tmp = p; p = q; q = tmp; }
        float apq = A[p][q];
        float c = 1.0f, s = 0.0f;
        if (fabsf(apq) > 1e-20f) {
          float theta = (A[q][q] - A[p][p]) / (2.0f * apq);
          float tt = copysignf(1.0f, theta) /
                     (fabsf(theta) + sqrtf(theta * theta + 1.0f));
          c = rsqrtf(tt * tt + 1.0f);
          s = tt * c;
        }
        rp[tid] = p; rq[tid] = q; rc[tid] = c; rs[tid] = s;
      }
      __syncthreads();
      // column rotations A <- A*J, V <- V*J : thread (rr, gg)
      #pragma unroll
      for (int i = 0; i < 8; ++i) {
        const int j = gg * 8 + i;
        const int p = rp[j], q = rq[j];
        const float c = rc[j], s = rs[j];
        float ap = A[rr][p], aq = A[rr][q];
        A[rr][p] = c * ap - s * aq;
        A[rr][q] = s * ap + c * aq;
        float vp = V[rr][p], vq = V[rr][q];
        V[rr][p] = c * vp - s * vq;
        V[rr][q] = s * vp + c * vq;
      }
      __syncthreads();
      // row rotations A <- J^T*A : thread (rr=column, gg)
      #pragma unroll
      for (int i = 0; i < 8; ++i) {
        const int j = gg * 8 + i;
        const int p = rp[j], q = rq[j];
        const float c = rc[j], s = rs[j];
        float ap = A[p][rr], aq = A[q][rr];
        A[p][rr] = c * ap - s * aq;
        A[q][rr] = s * ap + c * aq;
      }
      __syncthreads();
    }
  }

  // eigenvalues on the diagonal; rank-sort ascending (eigh convention)
  if (tid < 64) wvals[tid] = A[tid][tid];
  __syncthreads();
  if (tid < 64) {
    const float wv = wvals[tid];
    int rank = 0;
    for (int j = 0; j < 64; ++j) {
      float o = wvals[j];
      rank += (o < wv || (o == wv && j < tid)) ? 1 : 0;
    }
    colOf[rank] = tid;          // column holding sorted eigenvalue #rank
    wperm[tid]  = Wpt[rank];    // W_pt permuted to original column order
  }
  __syncthreads();

  // out[b, i*nh + head] = w_sorted[i] * sum_j V_sorted[i,j] * W_pt[j] + b_pt
  //                     = wvals[colOf[i]] * sum_m V[i][m]*Wpt[rank[m]] + b_pt
  if (tid < 64) {
    float dot = 0.0f;
    for (int m = 0; m < 64; ++m) dot += V[tid][m] * wperm[m];
    const float ev = wvals[colOf[tid]];
    const int bIdx = mat >> 2;
    const int hIdx = mat & 3;
    out[bIdx * H_ + tid * NH_ + hIdx] = ev * dot + bpt[0];
  }
}

// ---------------------------------------------------------------------------
extern "C" void kernel_launch(void* const* d_in, const int* in_sizes, int n_in,
                              void* d_out, int out_size, void* d_ws, size_t ws_size,
                              hipStream_t stream) {
  (void)in_sizes; (void)n_in; (void)out_size; (void)ws_size;
  const float* x   = (const float*)d_in[0];   // [16,8192,256]
  const float* Wff = (const float*)d_in[1];   // [256,256]
  const float* bff = (const float*)d_in[2];   // [256]
  const float* Wpt = (const float*)d_in[3];   // [64,1]
  const float* bpt = (const float*)d_in[4];   // [1]
  float* out = (float*)d_out;                 // [16,256]
  float* cov = (float*)d_ws;                  // needs 16*4*64*64*4 = 1 MB

  const int ncov = B_ * NH_ * HS_ * HS_;
  k_zero<<<(ncov + 255) / 256, 256, 0, stream>>>(cov, ncov);
  k_ff_cov<<<dim3(S_ / MT_, B_), 256, 0, stream>>>(x, Wff, bff, cov);
  k_eigh_out<<<B_ * NH_, 256, 0, stream>>>(cov, Wpt, bpt, out);
}
